// CoAttention_58128087384200
// MI455X (gfx1250) — matrix operand order repaired and checked by reading, not compiled, not run on hardware
//
#include <hip/hip_runtime.h>

// Problem sizes (fixed by the reference)
#define BB 32
#define TT 1024
#define II 512
#define DD 512

typedef __bf16 bf16;
typedef __attribute__((ext_vector_type(16))) __bf16 v16bf;
typedef __attribute__((ext_vector_type(8)))  __bf16 v8bf;
typedef __attribute__((ext_vector_type(4)))  __bf16 v4bf;
typedef __attribute__((ext_vector_type(8)))  float  v8f;
typedef __attribute__((ext_vector_type(4)))  float  v4f;
typedef __attribute__((ext_vector_type(2)))  int    v2i;

// A-slab staging: 32 rows x 32 k of bf16, row stride padded to 80 bytes
// (20 banks) so per-lane 16B fragment reads spread over all 64 LDS banks.
#define ROWB 80
#define SLAB (32 * ROWB)

#if defined(__HIP_DEVICE_COMPILE__) && __has_builtin(__builtin_amdgcn_global_load_async_to_lds_b64)
#define HAVE_ASYNC 1
#else
#define HAVE_ASYNC 0
#endif

__device__ __forceinline__ void stage_b64(const bf16* g, char* l)
{
#if HAVE_ASYNC
    __builtin_amdgcn_global_load_async_to_lds_b64(
        (__attribute__((address_space(1))) v2i*)(void*)const_cast<bf16*>(g),
        (__attribute__((address_space(3))) v2i*)(void*)l, 0, 0);
#else
    *(unsigned long long*)l = *(const unsigned long long*)g;
#endif
}

__device__ __forceinline__ void wait_async_1()
{
#if HAVE_ASYNC
 #if __has_builtin(__builtin_amdgcn_s_wait_asynccnt)
    __builtin_amdgcn_s_wait_asynccnt((short)1);
 #else
    asm volatile("s_wait_asynccnt 0x1" ::: "memory");
 #endif
#endif
}

__device__ __forceinline__ void wait_async_0()
{
#if HAVE_ASYNC
 #if __has_builtin(__builtin_amdgcn_s_wait_asynccnt)
    __builtin_amdgcn_s_wait_asynccnt((short)0);
 #else
    asm volatile("s_wait_asynccnt 0x0" ::: "memory");
 #endif
#endif
}

// Read one 16x32 bf16 A fragment from an LDS slab (per-lane layout:
// lane holds row lane&15; K chunk at (lane>=16)*16, contiguous).
__device__ __forceinline__ v16bf lds_a_frag(const char* slab, int tile_row0,
                                            int l15, int half)
{
    const char* p = slab + (tile_row0 + l15) * ROWB + half * 32;
    v8bf lo = *(const v8bf*)p;
    v8bf hi = *(const v8bf*)(p + 16);
    v16bf f;
#pragma unroll
    for (int e = 0; e < 8; e++) { f[e] = lo[e]; f[e + 8] = hi[e]; }
    return f;
}

// Read one 32x16 bf16 B fragment from a K-contiguous column in global memory
// (lane = column lane&15; K chunk at (lane>=16)*16).
__device__ __forceinline__ v16bf glb_b_frag(const bf16* bcol)
{
    v8bf lo = *(const v8bf*)(bcol);
    v8bf hi = *(const v8bf*)(bcol + 8);
    v16bf f;
#pragma unroll
    for (int e = 0; e < 8; e++) { f[e] = lo[e]; f[e + 8] = hi[e]; }
    return f;
}

// ---------------------------------------------------------------------------
// prep_text: v1[b,t] = text[b,t,:]·W1 ; a_bf[b,t,d] = bf16(text[b,t,d]*W3[d])
// ---------------------------------------------------------------------------
__global__ void __launch_bounds__(256)
prep_text_k(const float* __restrict__ text, const float* __restrict__ W1,
            const float* __restrict__ W3, bf16* __restrict__ a_out,
            float* __restrict__ v1_out)
{
    int wave = threadIdx.x >> 5, lane = threadIdx.x & 31;
    int row  = blockIdx.x * 8 + wave;                 // [0, B*T)
    const float* src = text + (size_t)row * DD;
    bf16* dst = a_out + (size_t)row * DD;
    int d0 = lane * 16;
    float s = 0.f;
#pragma unroll
    for (int j = 0; j < 16; j += 4) {
        v4f x  = *(const v4f*)(src + d0 + j);
        v4f w1 = *(const v4f*)(W1  + d0 + j);
        v4f w3 = *(const v4f*)(W3  + d0 + j);
        s += x[0]*w1[0] + x[1]*w1[1] + x[2]*w1[2] + x[3]*w1[3];
        v4bf o;
        o[0] = (bf16)(x[0]*w3[0]); o[1] = (bf16)(x[1]*w3[1]);
        o[2] = (bf16)(x[2]*w3[2]); o[3] = (bf16)(x[3]*w3[3]);
        *(v4bf*)(dst + d0 + j) = o;
    }
#pragma unroll
    for (int m = 16; m >= 1; m >>= 1) s += __shfl_xor(s, m, 32);
    if (lane == 0) v1_out[row] = s;
}

// ---------------------------------------------------------------------------
// prep_img: v2[b,i] = img·W2 ; img_bf ([B,I,D]) ; imgT ([B,D,I])
// ---------------------------------------------------------------------------
__global__ void __launch_bounds__(256)
prep_img_k(const float* __restrict__ img, const float* __restrict__ W2,
           bf16* __restrict__ img_out, bf16* __restrict__ imgT_out,
           float* __restrict__ v2_out)
{
    int wave = threadIdx.x >> 5, lane = threadIdx.x & 31;
    int row  = blockIdx.x * 8 + wave;                 // [0, B*I)
    int b = row / II, i = row - b * II;
    const float* src = img + (size_t)row * DD;
    bf16* dst = img_out + (size_t)row * DD;
    int d0 = lane * 16;
    float s = 0.f;
#pragma unroll
    for (int j = 0; j < 16; j += 4) {
        v4f x  = *(const v4f*)(src + d0 + j);
        v4f w2 = *(const v4f*)(W2  + d0 + j);
        s += x[0]*w2[0] + x[1]*w2[1] + x[2]*w2[2] + x[3]*w2[3];
        v4bf o;
        o[0] = (bf16)x[0]; o[1] = (bf16)x[1]; o[2] = (bf16)x[2]; o[3] = (bf16)x[3];
        *(v4bf*)(dst + d0 + j) = o;
#pragma unroll
        for (int e = 0; e < 4; e++)
            imgT_out[((size_t)b * DD + (d0 + j + e)) * II + i] = o[e];
    }
#pragma unroll
    for (int m = 16; m >= 1; m >>= 1) s += __shfl_xor(s, m, 32);
    if (lane == 0) v2_out[row] = s;
}

// ---------------------------------------------------------------------------
// Scores GEMM + softmax over I.
// Block: one batch b, 32 t-rows, full I=512.  8 waves; each wave computes a
// 2(t) x 4(i) grid of 16x16 WMMA tiles.  The shared 32x32 A slab is double-
// buffered in LDS via async global->LDS copies; B columns are K-contiguous
// global loads.
// ---------------------------------------------------------------------------
__global__ void __launch_bounds__(256)
k_scores_softmax(const bf16* __restrict__ a_bf, const bf16* __restrict__ img_bf,
                 const float* __restrict__ v1, const float* __restrict__ v2,
                 const float* __restrict__ bias, bf16* __restrict__ Aimg,
                 float* __restrict__ mrow)
{
    __shared__ float vbuf[32 * II];                       // 64 KB
    __shared__ __align__(16) char ldsA[2 * SLAB];         // 5 KB

    int b    = blockIdx.x >> 5;                           // T/32 = 32 blocks
    int t0   = (blockIdx.x & 31) << 5;
    int wave = threadIdx.x >> 5, lane = threadIdx.x & 31;
    int half = lane >> 4, l15 = lane & 15;

    const bf16* Abase = a_bf + ((size_t)(b * TT + t0)) * DD;
    int crow = threadIdx.x >> 3, cchunk = threadIdx.x & 7;     // slab copy job
    const bf16* gsrc = Abase + (size_t)crow * DD + cchunk * 4; // b64 per thread
    char* ldst = ldsA + crow * ROWB + cchunk * 8;

    v8f acc[2][4] = {};
    stage_b64(gsrc, ldst);                                 // k0 = 0 -> buf 0
    int cur = 0;

    for (int k0 = 0; k0 < DD; k0 += 32) {
        if (k0 + 32 < DD) {
            stage_b64(gsrc + (k0 + 32), ldst + (cur ^ 1) * SLAB);
            wait_async_1();
        } else {
            wait_async_0();
        }
        __syncthreads();

        const char* abuf = ldsA + cur * SLAB;
        v16bf af0 = lds_a_frag(abuf, 0,  l15, half);
        v16bf af1 = lds_a_frag(abuf, 16, l15, half);

#pragma unroll
        for (int ii = 0; ii < 4; ii++) {
            int i = (wave << 6) + (ii << 4) + l15;
            const bf16* bcol = img_bf + ((size_t)(b * II + i)) * DD + k0 + (half << 4);
            __builtin_prefetch(bcol + 32, 0, 0);
            v16bf bfr = glb_b_frag(bcol);
            acc[0][ii] = __builtin_amdgcn_wmma_f32_16x16x32_bf16(
                false, af0, false, bfr, (short)0, acc[0][ii], false, false);
            acc[1][ii] = __builtin_amdgcn_wmma_f32_16x16x32_bf16(
                false, af1, false, bfr, (short)0, acc[1][ii], false, false);
        }
        __syncthreads();                                   // WAR on slab buffer
        cur ^= 1;
    }

    // v = acc + v1[t] + v2[i] + bias  -> LDS
    float vb = bias[0];
#pragma unroll
    for (int ti = 0; ti < 2; ti++)
#pragma unroll
        for (int ii = 0; ii < 4; ii++) {
            int i = (wave << 6) + (ii << 4) + l15;
            float addi = v2[b * II + i] + vb;
#pragma unroll
            for (int r = 0; r < 8; r++) {
                int M = ti * 16 + r + (half << 3);
                vbuf[M * II + i] = acc[ti][ii][r] + v1[b * TT + t0 + M] + addi;
            }
        }
    __syncthreads();

    // Softmax over I: 8 threads per row, 64 elements per thread.
    int row = threadIdx.x >> 3, sub = threadIdx.x & 7;
    float* vr = vbuf + row * II + sub * 64;
    float mx = -3.4e38f;
#pragma unroll
    for (int j = 0; j < 64; j++) mx = fmaxf(mx, vr[j]);
#pragma unroll
    for (int m = 4; m >= 1; m >>= 1) mx = fmaxf(mx, __shfl_xor(mx, m, 32));
    if (sub == 0) mrow[b * TT + t0 + row] = mx;

    float ev[64], sum = 0.f;
#pragma unroll
    for (int j = 0; j < 64; j++) { ev[j] = __expf(vr[j] - mx); sum += ev[j]; }
#pragma unroll
    for (int m = 4; m >= 1; m >>= 1) sum += __shfl_xor(sum, m, 32);
    float inv = 1.f / sum;

    bf16* dst = Aimg + ((size_t)(b * TT + t0 + row)) * II + sub * 64;
#pragma unroll
    for (int j = 0; j < 64; j++) dst[j] = (bf16)(ev[j] * inv);
}

// ---------------------------------------------------------------------------
// A_text = softmax_T(m[b,:]) ; text_re[b,d] = sum_t A_text[t]*text[b,t,d]
// ---------------------------------------------------------------------------
__global__ void __launch_bounds__(256)
k_text_att(const float* __restrict__ mrow, const float* __restrict__ text,
           float* __restrict__ text_re)
{
    __shared__ float red[8];
    __shared__ float at[TT];
    int b = blockIdx.x, tid = threadIdx.x;

    float mv[4], lm = -3.4e38f;
#pragma unroll
    for (int j = 0; j < 4; j++) {
        mv[j] = mrow[b * TT + tid * 4 + j];
        lm = fmaxf(lm, mv[j]);
    }
#pragma unroll
    for (int m = 16; m >= 1; m >>= 1) lm = fmaxf(lm, __shfl_xor(lm, m, 32));
    if ((tid & 31) == 0) red[tid >> 5] = lm;
    __syncthreads();
    if (tid == 0) { float g = red[0]; for (int w = 1; w < 8; w++) g = fmaxf(g, red[w]); red[0] = g; }
    __syncthreads();
    float gm = red[0];
    __syncthreads();

    float ev[4], ls = 0.f;
#pragma unroll
    for (int j = 0; j < 4; j++) { ev[j] = __expf(mv[j] - gm); ls += ev[j]; }
#pragma unroll
    for (int m = 16; m >= 1; m >>= 1) ls += __shfl_xor(ls, m, 32);
    if ((tid & 31) == 0) red[tid >> 5] = ls;
    __syncthreads();
    if (tid == 0) { float g = 0.f; for (int w = 0; w < 8; w++) g += red[w]; red[0] = g; }
    __syncthreads();
    float inv = 1.f / red[0];
#pragma unroll
    for (int j = 0; j < 4; j++) at[tid * 4 + j] = ev[j] * inv;
    __syncthreads();

    for (int d = tid; d < DD; d += 256) {
        const float* col = text + (size_t)b * TT * DD + d;
        float acc = 0.f;
        for (int t = 0; t < TT; t++) acc += at[t] * col[(size_t)t * DD];
        text_re[b * DD + d] = acc;
    }
}

// ---------------------------------------------------------------------------
// img_re = A_img · img (K = I), then assemble G = [text | img_re |
// text*img_re | text*text_re] straight from the accumulators.
// Same tiling / staging as the scores GEMM.
// ---------------------------------------------------------------------------
__global__ void __launch_bounds__(256)
k_imgre_assemble(const bf16* __restrict__ Aimg, const bf16* __restrict__ imgT,
                 const float* __restrict__ text, const float* __restrict__ text_re,
                 float* __restrict__ out)
{
    __shared__ __align__(16) char ldsA[2 * SLAB];

    int b    = blockIdx.x >> 5;
    int t0   = (blockIdx.x & 31) << 5;
    int wave = threadIdx.x >> 5, lane = threadIdx.x & 31;
    int half = lane >> 4, l15 = lane & 15;

    const bf16* Abase = Aimg + ((size_t)(b * TT + t0)) * II;
    int crow = threadIdx.x >> 3, cchunk = threadIdx.x & 7;
    const bf16* gsrc = Abase + (size_t)crow * II + cchunk * 4;
    char* ldst = ldsA + crow * ROWB + cchunk * 8;

    v8f acc[2][4] = {};
    stage_b64(gsrc, ldst);
    int cur = 0;

    for (int k0 = 0; k0 < II; k0 += 32) {
        if (k0 + 32 < II) {
            stage_b64(gsrc + (k0 + 32), ldst + (cur ^ 1) * SLAB);
            wait_async_1();
        } else {
            wait_async_0();
        }
        __syncthreads();

        const char* abuf = ldsA + cur * SLAB;
        v16bf af0 = lds_a_frag(abuf, 0,  l15, half);
        v16bf af1 = lds_a_frag(abuf, 16, l15, half);

#pragma unroll
        for (int ii = 0; ii < 4; ii++) {
            int d = (wave << 6) + (ii << 4) + l15;
            const bf16* bcol = imgT + ((size_t)(b * DD + d)) * II + k0 + (half << 4);
            __builtin_prefetch(bcol + 32, 0, 0);
            v16bf bfr = glb_b_frag(bcol);
            acc[0][ii] = __builtin_amdgcn_wmma_f32_16x16x32_bf16(
                false, af0, false, bfr, (short)0, acc[0][ii], false, false);
            acc[1][ii] = __builtin_amdgcn_wmma_f32_16x16x32_bf16(
                false, af1, false, bfr, (short)0, acc[1][ii], false, false);
        }
        __syncthreads();
        cur ^= 1;
    }

#pragma unroll
    for (int ti = 0; ti < 2; ti++)
#pragma unroll
        for (int ii = 0; ii < 4; ii++) {
            int d = (wave << 6) + (ii << 4) + l15;
            float tre = text_re[b * DD + d];
#pragma unroll
            for (int r = 0; r < 8; r++) {
                int t = t0 + ti * 16 + r + (half << 3);
                float ire = acc[ti][ii][r];
                float tx  = text[((size_t)(b * TT + t)) * DD + d];
                size_t base = ((size_t)(b * TT + t)) * (4 * DD);
                out[base + d]          = tx;
                out[base + DD + d]     = ire;
                out[base + 2 * DD + d] = tx * ire;
                out[base + 3 * DD + d] = tx * tre;
            }
        }
}

// ---------------------------------------------------------------------------
extern "C" void kernel_launch(void* const* d_in, const int* in_sizes, int n_in,
                              void* d_out, int out_size, void* d_ws, size_t ws_size,
                              hipStream_t stream)
{
    const float* text = (const float*)d_in[0];
    const float* img  = (const float*)d_in[1];
    const float* W1   = (const float*)d_in[2];
    const float* W2   = (const float*)d_in[3];
    const float* W3   = (const float*)d_in[4];
    const float* bias = (const float*)d_in[5];
    float* out = (float*)d_out;

    char* ws = (char*)d_ws;
    size_t off = 0;
    bf16* a_bf   = (bf16*)(ws + off); off += (size_t)BB * TT * DD * 2;  // 32 MB
    bf16* img_bf = (bf16*)(ws + off); off += (size_t)BB * II * DD * 2;  // 16 MB
    bf16* imgT   = (bf16*)(ws + off); off += (size_t)BB * DD * II * 2;  // 16 MB
    bf16* Aimg   = (bf16*)(ws + off); off += (size_t)BB * TT * II * 2;  // 32 MB
    float* v1    = (float*)(ws + off); off += (size_t)BB * TT * 4;
    float* v2    = (float*)(ws + off); off += (size_t)BB * II * 4;
    float* mrow  = (float*)(ws + off); off += (size_t)BB * TT * 4;
    float* tre   = (float*)(ws + off); off += (size_t)BB * DD * 4;

    prep_text_k<<<BB * TT / 8, 256, 0, stream>>>(text, W1, W3, a_bf, v1);
    prep_img_k<<<BB * II / 8, 256, 0, stream>>>(img, W2, img_bf, imgT, v2);
    k_scores_softmax<<<BB * (TT / 32), 256, 0, stream>>>(a_bf, img_bf, v1, v2,
                                                         bias, Aimg, mrow);
    k_text_att<<<BB, 256, 0, stream>>>(mrow, text, tre);
    k_imgre_assemble<<<BB * (TT / 32), 256, 0, stream>>>(Aimg, imgT, text, tre, out);
}
